// QAttention_61718680043569
// MI455X (gfx1250) — 8-bit matrix operand order repaired and checked by reading, compiled, not run on hardware
//
#include <hip/hip_runtime.h>


typedef int i32x8 __attribute__((ext_vector_type(8)));

__device__ __forceinline__ float load_scale(const unsigned* slot) {
  float amax = __uint_as_float(*slot);
  return fmaxf(amax * (1.0f / 127.0f), 1e-8f);
}

__device__ __forceinline__ signed char quant1(float x, float inv) {
  int q = __float2int_rn(x * inv);
  q = q > 127 ? 127 : q;
  q = q < -128 ? -128 : q;
  return (signed char)q;
}

__device__ __forceinline__ void async_b128(const void* g, void* lds) {
  asm volatile("global_load_async_to_lds_b128 %0, %1, off"
               :: "v"((unsigned)(size_t)lds), "v"(g)
               : "memory");
}
__device__ __forceinline__ void wait_async0() {
  asm volatile("s_wait_asynccnt 0x0" ::: "memory");
}

__device__ __forceinline__ i32x8 lds_afrag(const signed char* p, int lane) {
  const signed char* q = p + (lane & 15) * 64 + ((lane >> 4) << 3);
  i32x8 f;
#pragma unroll
  for (int c = 0; c < 4; ++c) {
    int2 d = *(const int2*)(q + c * 16);
    f[2 * c] = d.x;
    f[2 * c + 1] = d.y;
  }
  return f;
}

__device__ __forceinline__ i32x8 lds_bfrag(const signed char* p, int lane) {
  const signed char* q = p + (lane & 15) * 64 + ((lane >> 4) << 3);
  i32x8 f;
#pragma unroll
  for (int j = 0; j < 8; ++j) {
    const int k = ((j >> 1) << 4) + ((j & 1) << 2);
    f[j] = *(const int*)(q + k);
  }
  return f;
}

__global__ void zero_slots_kernel(unsigned* slots) {
  if (threadIdx.x < 8) slots[threadIdx.x] = 0u;
}

__global__ void absmax_kernel(const float* __restrict__ x, int n, unsigned* slot) {
  float m = 0.f;
  for (int i = blockIdx.x * blockDim.x + threadIdx.x; i < n; i += gridDim.x * blockDim.x)
    m = fmaxf(m, fabsf(x[i]));
#pragma unroll
  for (int mask = 16; mask > 0; mask >>= 1) m = fmaxf(m, __shfl_xor(m, mask, 32));
  if ((threadIdx.x & 31) == 0) atomicMax(slot, __float_as_uint(m));
}

__global__ void quant_kernel(const float* __restrict__ x, int n,
                             const unsigned* __restrict__ slot,
                             signed char* __restrict__ out) {
  const float inv = 1.0f / load_scale(slot);
  for (int i = blockIdx.x * blockDim.x + threadIdx.x; i < n; i += gridDim.x * blockDim.x)
    out[i] = quant1(x[i], inv);
}

__global__ void absmax_qkv_kernel(const float* __restrict__ qkv, unsigned* slots3) {
  float m0 = 0.f, m1 = 0.f, m2 = 0.f;
  const int n = 8192 * 2304;
  for (int i = blockIdx.x * blockDim.x + threadIdx.x; i < n; i += gridDim.x * blockDim.x) {
    const int s = (i % 2304) / 768;
    const float v = fabsf(qkv[i]);
    m0 = (s == 0) ? fmaxf(m0, v) : m0;
    m1 = (s == 1) ? fmaxf(m1, v) : m1;
    m2 = (s == 2) ? fmaxf(m2, v) : m2;
  }
#pragma unroll
  for (int mask = 16; mask > 0; mask >>= 1) {
    m0 = fmaxf(m0, __shfl_xor(m0, mask, 32));
    m1 = fmaxf(m1, __shfl_xor(m1, mask, 32));
    m2 = fmaxf(m2, __shfl_xor(m2, mask, 32));
  }
  if ((threadIdx.x & 31) == 0) {
    atomicMax(slots3 + 0, __float_as_uint(m0));
    atomicMax(slots3 + 1, __float_as_uint(m1));
    atomicMax(slots3 + 2, __float_as_uint(m2));
  }
}

__global__ void qt_qkv_kernel(const float* __restrict__ qkv, const unsigned* __restrict__ slots,
                              signed char* __restrict__ q8, signed char* __restrict__ k8,
                              signed char* __restrict__ v8t) {
  const float iq = 1.0f / load_scale(slots + 3);
  const float ik = 1.0f / load_scale(slots + 4);
  const float iv = 1.0f / load_scale(slots + 5);
  const int total = 8192 * 2304;
  for (int i = blockIdx.x * blockDim.x + threadIdx.x; i < total; i += gridDim.x * blockDim.x) {
    const int m = i / 2304, col = i % 2304;
    const int s = col / 768, r = col % 768;
    const int h = r >> 6, d = r & 63;
    const int b = m >> 10, n = m & 1023;
    const long bh = (long)b * 12 + h;
    const float v = qkv[i];
    if (s == 0)      q8[((bh * 1024 + n) << 6) + d] = quant1(v, iq);
    else if (s == 1) k8[((bh * 1024 + n) << 6) + d] = quant1(v, ik);
    else             v8t[((bh * 64 + d) << 10) + n] = quant1(v, iv);
  }
}

__global__ __launch_bounds__(256) void gemm_i8_kernel(
    const signed char* __restrict__ A, const signed char* __restrict__ Bt,
    float* __restrict__ C, const float* __restrict__ bias,
    const unsigned* __restrict__ sA, const unsigned* __restrict__ sB,
    int M, int N, int K) {
  __shared__ signed char As[2][128 * 64];
  __shared__ signed char Bs[2][128 * 64];
  const int tid = threadIdx.x, lane = tid & 31, wave = tid >> 5;
  const int wm = wave >> 2, wn = wave & 3;
  const int m0 = blockIdx.y * 128, n0 = blockIdx.x * 128;
  const int row = tid >> 1, off = (tid & 1) * 32;

  i32x8 acc[4][2];
  const i32x8 zero = {0, 0, 0, 0, 0, 0, 0, 0};
#pragma unroll
  for (int t = 0; t < 4; ++t)
#pragma unroll
    for (int u = 0; u < 2; ++u) acc[t][u] = zero;

  {
    const signed char* ga = A + (long)(m0 + row) * K + off;
    const signed char* gb = Bt + (long)(n0 + row) * K + off;
    async_b128(ga,      &As[0][row * 64 + off]);
    async_b128(ga + 16, &As[0][row * 64 + off + 16]);
    async_b128(gb,      &Bs[0][row * 64 + off]);
    async_b128(gb + 16, &Bs[0][row * 64 + off + 16]);
  }
  wait_async0();
  __syncthreads();

  const int nk = K >> 6;
  for (int s = 0; s < nk; ++s) {
    const int buf = s & 1;
    if (s + 1 < nk) {
      const int ks = (s + 1) << 6, nb = buf ^ 1;
      const signed char* ga = A + (long)(m0 + row) * K + ks + off;
      const signed char* gb = Bt + (long)(n0 + row) * K + ks + off;
      async_b128(ga,      &As[nb][row * 64 + off]);
      async_b128(ga + 16, &As[nb][row * 64 + off + 16]);
      async_b128(gb,      &Bs[nb][row * 64 + off]);
      async_b128(gb + 16, &Bs[nb][row * 64 + off + 16]);
    }
    if (s + 2 < nk) {
      __builtin_prefetch(A + (long)(m0 + row) * K + ((s + 2) << 6) + off, 0, 1);
      __builtin_prefetch(Bt + (long)(n0 + row) * K + ((s + 2) << 6) + off, 0, 1);
    }
    i32x8 a[4], b[2];
#pragma unroll
    for (int t = 0; t < 4; ++t) a[t] = lds_afrag(&As[buf][(wm * 64 + t * 16) * 64], lane);
#pragma unroll
    for (int u = 0; u < 2; ++u) b[u] = lds_bfrag(&Bs[buf][(wn * 32 + u * 16) * 64], lane);
#pragma unroll
    for (int t = 0; t < 4; ++t)
#pragma unroll
      for (int u = 0; u < 2; ++u)
        acc[t][u] = __builtin_amdgcn_wmma_i32_16x16x64_iu8(true, a[t], true, b[u],
                                                           acc[t][u], false, false);
    wait_async0();
    __syncthreads();
  }

  const float scale = load_scale(sA) * load_scale(sB);
#pragma unroll
  for (int t = 0; t < 4; ++t)
#pragma unroll
    for (int u = 0; u < 2; ++u) {
      const int mr = m0 + wm * 64 + t * 16 + ((lane >> 4) << 3);
      const int nc = n0 + wn * 32 + u * 16 + (lane & 15);
      const float bv = bias ? bias[nc] : 0.f;
#pragma unroll
      for (int i = 0; i < 8; ++i)
        C[(long)(mr + i) * N + nc] = (float)acc[t][u][i] * scale + bv;
    }
}

__global__ __launch_bounds__(256) void flash_attn_kernel(
    const signed char* __restrict__ q8, const signed char* __restrict__ k8,
    const signed char* __restrict__ v8t, float* __restrict__ O,
    const unsigned* __restrict__ slots) {
  __shared__ signed char Ks[64 * 64];
  __shared__ signed char Vs[64 * 64];
  __shared__ signed char Ps[8][16 * 64];
  const int tid = threadIdx.x, lane = tid & 31, wave = tid >> 5;
  const int bh = blockIdx.x, b = bh / 12, h = bh % 12;
  const int nb = blockIdx.y * 128;
  const float sq = load_scale(slots + 3), sk = load_scale(slots + 4), sv = load_scale(slots + 5);
  const float qk_scale = sq * sk * 0.125f;
  const float pv_scale = sv * (1.0f / 127.0f);
  const i32x8 zero = {0, 0, 0, 0, 0, 0, 0, 0};

  i32x8 qa;
  {
    const signed char* qp =
        q8 + ((long)bh * 1024 + nb + wave * 16 + (lane & 15)) * 64 + ((lane >> 4) << 3);
#pragma unroll
    for (int c = 0; c < 4; ++c) {
      int2 d = *(const int2*)(qp + c * 16);
      qa[2 * c] = d.x;
      qa[2 * c + 1] = d.y;
    }
  }

  float o[4][8], mrow[8], lrow[8];
#pragma unroll
  for (int t = 0; t < 4; ++t)
#pragma unroll
    for (int i = 0; i < 8; ++i) o[t][i] = 0.f;
#pragma unroll
  for (int i = 0; i < 8; ++i) { mrow[i] = -1e30f; lrow[i] = 0.f; }

  for (int j0 = 0; j0 < 1024; j0 += 64) {
    async_b128(k8 + ((long)bh * 1024 + j0) * 64 + tid * 16, Ks + tid * 16);
    {
      const int d = tid >> 2, ch = (tid & 3) << 4;
      async_b128(v8t + ((long)bh * 64 + d) * 1024 + j0 + ch, Vs + d * 64 + ch);
    }
    wait_async0();
    __syncthreads();

    i32x8 st[4];
#pragma unroll
    for (int t = 0; t < 4; ++t) {
      i32x8 kb = lds_bfrag(&Ks[t * 16 * 64], lane);
      st[t] = __builtin_amdgcn_wmma_i32_16x16x64_iu8(true, qa, true, kb, zero, false, false);
    }

    float p[4][8], alpha[8];
#pragma unroll
    for (int r = 0; r < 8; ++r) {
      float mx = mrow[r];
#pragma unroll
      for (int t = 0; t < 4; ++t) mx = fmaxf(mx, (float)st[t][r] * qk_scale);
#pragma unroll
      for (int mask = 1; mask <= 8; mask <<= 1) mx = fmaxf(mx, __shfl_xor(mx, mask, 32));
      alpha[r] = __expf(mrow[r] - mx);
      mrow[r] = mx;
      float rs = 0.f;
#pragma unroll
      for (int t = 0; t < 4; ++t) {
        const float pv = __expf((float)st[t][r] * qk_scale - mx);
        p[t][r] = pv;
        rs += pv;
      }
#pragma unroll
      for (int mask = 1; mask <= 8; mask <<= 1) rs += __shfl_xor(rs, mask, 32);
      lrow[r] = lrow[r] * alpha[r] + rs;
    }

    signed char* pw = Ps[wave];
#pragma unroll
    for (int r = 0; r < 8; ++r) {
      const int prow = ((lane >> 4) << 3) + r;
#pragma unroll
      for (int t = 0; t < 4; ++t) {
        int qv = __float2int_rn(p[t][r] * 127.0f);
        qv = qv > 127 ? 127 : (qv < 0 ? 0 : qv);
        pw[prow * 64 + t * 16 + (lane & 15)] = (signed char)qv;
      }
    }

#pragma unroll
    for (int t = 0; t < 4; ++t)
#pragma unroll
      for (int i = 0; i < 8; ++i) o[t][i] *= alpha[i];

    i32x8 pa = lds_afrag(pw, lane);
#pragma unroll
    for (int t = 0; t < 4; ++t) {
      i32x8 vb = lds_bfrag(&Vs[t * 16 * 64], lane);
      i32x8 d = __builtin_amdgcn_wmma_i32_16x16x64_iu8(true, pa, true, vb, zero, false, false);
#pragma unroll
      for (int i = 0; i < 8; ++i) o[t][i] += (float)d[i] * pv_scale;
    }
    __syncthreads();
  }

  const int nrow = nb + wave * 16 + ((lane >> 4) << 3);
#pragma unroll
  for (int i = 0; i < 8; ++i) {
    const float invl = 1.0f / lrow[i];
#pragma unroll
    for (int t = 0; t < 4; ++t)
      O[((long)b * 1024 + nrow + i) * 768 + h * 64 + t * 16 + (lane & 15)] = o[t][i] * invl;
  }
}

extern "C" void kernel_launch(void* const* d_in, const int* in_sizes, int n_in,
                              void* d_out, int out_size, void* d_ws, size_t ws_size,
                              hipStream_t stream) {
  (void)in_sizes; (void)n_in; (void)out_size; (void)ws_size;
  const float* x      = (const float*)d_in[0];
  const float* w_qkv  = (const float*)d_in[1];
  const float* b_qkv  = (const float*)d_in[2];
  const float* w_proj = (const float*)d_in[3];
  const float* b_proj = (const float*)d_in[4];
  float* out = (float*)d_out;

  char* ws = (char*)d_ws;
  unsigned* slots   = (unsigned*)ws;
  signed char* x8   = (signed char*)(ws + 256);
  signed char* wq8  = (signed char*)(ws + 6291712);
  signed char* wp8  = (signed char*)(ws + 8061184);
  float* qkv        = (float*)(ws + 8651008);
  signed char* q8   = (signed char*)(ws + 84148480);
  signed char* k8   = (signed char*)(ws + 90439936);
  signed char* v8t  = (signed char*)(ws + 96731392);
  float* Obuf       = qkv;
  signed char* o8   = x8;

  zero_slots_kernel<<<1, 32, 0, stream>>>(slots);

  absmax_kernel<<<512, 256, 0, stream>>>(x, 8192 * 768, slots + 0);
  absmax_kernel<<<256, 256, 0, stream>>>(w_qkv, 2304 * 768, slots + 1);
  absmax_kernel<<<128, 256, 0, stream>>>(w_proj, 768 * 768, slots + 2);
  quant_kernel<<<1024, 256, 0, stream>>>(x, 8192 * 768, slots + 0, x8);
  quant_kernel<<<512, 256, 0, stream>>>(w_qkv, 2304 * 768, slots + 1, wq8);
  quant_kernel<<<256, 256, 0, stream>>>(w_proj, 768 * 768, slots + 2, wp8);

  gemm_i8_kernel<<<dim3(2304 / 128, 8192 / 128), 256, 0, stream>>>(
      x8, wq8, qkv, b_qkv, slots + 0, slots + 1, 8192, 2304, 768);

  absmax_qkv_kernel<<<1024, 256, 0, stream>>>(qkv, slots + 3);
  qt_qkv_kernel<<<2048, 256, 0, stream>>>(qkv, slots, q8, k8, v8t);

  flash_attn_kernel<<<dim3(96, 8), 256, 0, stream>>>(q8, k8, v8t, Obuf, slots);

  absmax_kernel<<<512, 256, 0, stream>>>(Obuf, 8192 * 768, slots + 6);
  quant_kernel<<<1024, 256, 0, stream>>>(Obuf, 8192 * 768, slots + 6, o8);

  gemm_i8_kernel<<<dim3(768 / 128, 8192 / 128), 256, 0, stream>>>(
      o8, wp8, out, b_proj, slots + 6, slots + 2, 8192, 768, 768);
}
